// SNNController_62620623176344
// MI455X (gfx1250) — matrix operand order repaired and checked by reading, not compiled, not run on hardware
//
#include <hip/hip_runtime.h>

typedef __attribute__((ext_vector_type(16))) __bf16 v16bf;
typedef __attribute__((ext_vector_type(8)))  float  v8f;

#define DECAY  0.8f
#define THRESH 1.0f

#if defined(__has_builtin)
#if __has_builtin(__builtin_amdgcn_cvt_pk_bf16_f32)
#define HAS_HW_PKBF16 1
#endif
#endif

// Pack two fp32 -> packed bf16 dword, lo in [15:0].
__device__ __forceinline__ unsigned pack_bf16(float lo, float hi) {
#ifdef HAS_HW_PKBF16
  auto r = __builtin_amdgcn_cvt_pk_bf16_f32(lo, hi);   // v_cvt_pk_bf16_f32
  unsigned u;
  __builtin_memcpy(&u, &r, sizeof(u));
  return u;
#else
  // round-half-up then byte-perm: 3 VALU total (v_add, v_add, v_perm_b32)
  unsigned a = __float_as_uint(lo) + 0x8000u;
  unsigned b = __float_as_uint(hi) + 0x8000u;
  return __builtin_amdgcn_perm(b, a, 0x07060302u);
#endif
}

// Butterfly add across lanes with XOR pattern via ds_swizzle_b32 (1 DS op).
template <int MASK>
__device__ __forceinline__ float xor_add(float v) {
  int j = __builtin_amdgcn_ds_swizzle(__float_as_int(v), (MASK << 10) | 0x1f);
  return v + __int_as_float(j);
}

__global__ __launch_bounds__(256, 1)
void snn_fused_kernel(const float* __restrict__ X,    // [B,1024]
                      const float* __restrict__ W1,   // [64,1024]
                      const float* __restrict__ b1,   // [64]
                      const float* __restrict__ W2,   // [4,64]
                      const float* __restrict__ b2,   // [4]
                      const int*   __restrict__ tsp,  // scalar time_steps
                      float*       __restrict__ out)  // [B,4]
{
  // W1 transposed to [k][h] in bf16: 1024*64*2 = 128 KB (< 320 KB WGP LDS)
  __shared__ unsigned short w1t[64 * 1024];

  const int tid  = threadIdx.x;
  const int lane = tid & 31;
  const int wave = tid >> 5;

  // ---------------- Stage W1^T (bf16) into LDS ----------------
  {
    const int hp = tid & 31;       // h-pair index 0..31  (h0 = 2*hp)
    const int kb = tid >> 5;       // k-block 0..7, 128 k each
    const int h0 = hp * 2;
    const float* r0 = W1 + (size_t)h0 * 1024;
    const float* r1 = r0 + 1024;
    unsigned* w1t32 = (unsigned*)w1t;
    #pragma unroll 4
    for (int k = kb * 128; k < kb * 128 + 128; k += 4) {
      float4 a = *(const float4*)(r0 + k);
      float4 b = *(const float4*)(r1 + k);
      w1t32[((k + 0) * 64 + h0) >> 1] = pack_bf16(a.x, b.x);
      w1t32[((k + 1) * 64 + h0) >> 1] = pack_bf16(a.y, b.y);
      w1t32[((k + 2) * 64 + h0) >> 1] = pack_bf16(a.z, b.z);
      w1t32[((k + 3) * 64 + h0) >> 1] = pack_bf16(a.w, b.w);
    }
  }
  __syncthreads();

  // ---------------- Layer-1 GEMM via WMMA bf16 ----------------
  // Wave computes current1 tile: 16 batch rows x 64 hidden.
  const int rowBase = blockIdx.x * 128 + wave * 16;
  const int mrow    = rowBase + (lane & 15);
  const int koff    = (lane >> 4) << 3;          // 0 (lanes 0-15) or 8 (16-31)
  const float* arow = X + (size_t)mrow * 1024;

  v8f zero8 = {0.f, 0.f, 0.f, 0.f, 0.f, 0.f, 0.f, 0.f};
  v8f c[4];
  #pragma unroll
  for (int t = 0; t < 4; ++t) c[t] = zero8;

  #pragma unroll 2
  for (int k0 = 0; k0 < 1024; k0 += 32) {
    // stream-prefetch 1 KB ahead (speculative; dropped if unmapped)
    __builtin_prefetch(arow + k0 + 256, 0, 0);

    // A fragment (16x32 bf16): lanes<16 -> K {0..7,16..23}; lanes>=16 -> +8
    float4 a0 = *(const float4*)(arow + k0 + koff);
    float4 a1 = *(const float4*)(arow + k0 + koff + 4);
    float4 a2 = *(const float4*)(arow + k0 + koff + 16);
    float4 a3 = *(const float4*)(arow + k0 + koff + 20);
    union { v16bf v; unsigned u[8]; } A;
    A.u[0] = pack_bf16(a0.x, a0.y);
    A.u[1] = pack_bf16(a0.z, a0.w);
    A.u[2] = pack_bf16(a1.x, a1.y);
    A.u[3] = pack_bf16(a1.z, a1.w);
    A.u[4] = pack_bf16(a2.x, a2.y);
    A.u[5] = pack_bf16(a2.z, a2.w);
    A.u[6] = pack_bf16(a3.x, a3.y);
    A.u[7] = pack_bf16(a3.z, a3.w);

    // B fragments (32x16 bf16): lane -> N = 16*t + (lane&15); K in same order as A
    const unsigned short* bcol = w1t + (k0 + koff) * 64 + (lane & 15);
    #pragma unroll
    for (int t = 0; t < 4; ++t) {
      union { v16bf v; unsigned short u[16]; } Bf;
      #pragma unroll
      for (int s = 0; s < 16; ++s) Bf.u[s] = bcol[(s < 8 ? s : s + 8) * 64 + t * 16];
      c[t] = __builtin_amdgcn_wmma_f32_16x16x32_bf16(
                 false, A.v, false, Bf.v, (short)0, c[t], false, false);
    }
  }

  // ---------------- Fused LIF recurrence (registers only) ----------------
  // C layout: VGPR r -> batch row rowBase+r (lanes 0-15) / rowBase+r+8 (16-31),
  //           hidden h = 16*t + (lane&15).
  const int lhid = lane & 15;
  float b1v[4], w2v[4][4];
  #pragma unroll
  for (int t = 0; t < 4; ++t) {
    b1v[t] = b1[t * 16 + lhid];
    #pragma unroll
    for (int a = 0; a < 4; ++a) w2v[a][t] = W2[a * 64 + t * 16 + lhid];
  }
  const int   asel = lane & 3;        // action owned by this lane group
  const float b2v  = b2[asel];
  const int   T    = *tsp;

  float cur[4][8], mem1[4][8];
  #pragma unroll
  for (int t = 0; t < 4; ++t)
    #pragma unroll
    for (int r = 0; r < 8; ++r) { cur[t][r] = c[t][r] + b1v[t]; mem1[t][r] = 0.f; }

  float mem2[8], accv[8];
  #pragma unroll
  for (int r = 0; r < 8; ++r) { mem2[r] = 0.f; accv[r] = 0.f; }

  for (int step = 0; step < T; ++step) {
    #pragma unroll
    for (int r = 0; r < 8; ++r) {
      float sv[4];
      #pragma unroll
      for (int t = 0; t < 4; ++t) {
        float mm   = __builtin_fmaf(mem1[t][r], DECAY, cur[t][r]);
        bool  fire = (mm >= THRESH);
        sv[t]      = fire ? 1.0f : 0.0f;
        mem1[t][r] = fire ? 0.0f : mm;     // reset fired neurons
      }
      // per-lane weighted partials for all 4 actions (4 of 64 hidden each)
      float p0 = sv[0]*w2v[0][0] + sv[1]*w2v[0][1] + sv[2]*w2v[0][2] + sv[3]*w2v[0][3];
      float p1 = sv[0]*w2v[1][0] + sv[1]*w2v[1][1] + sv[2]*w2v[1][2] + sv[3]*w2v[1][3];
      float p2 = sv[0]*w2v[2][0] + sv[1]*w2v[2][1] + sv[2]*w2v[2][2] + sv[3]*w2v[2][3];
      float p3 = sv[0]*w2v[3][0] + sv[1]*w2v[3][1] + sv[2]*w2v[3][2] + sv[3]*w2v[3][3];
      // butterfly within groups of 4 lanes (stays inside each 16-lane half)
      p0 = xor_add<1>(p0);  p0 = xor_add<2>(p0);
      p1 = xor_add<1>(p1);  p1 = xor_add<2>(p1);
      p2 = xor_add<1>(p2);  p2 = xor_add<2>(p2);
      p3 = xor_add<1>(p3);  p3 = xor_add<2>(p3);
      // each lane keeps action (lane&3): invariant under xor {4,8}
      float p = (asel == 0) ? p0 : (asel == 1) ? p1 : (asel == 2) ? p2 : p3;
      p = xor_add<4>(p);
      p = xor_add<8>(p);
      // current2 = p + b2 ; LIF on layer 2
      float m2   = __builtin_fmaf(mem2[r], DECAY, p + b2v);
      bool  fire = (m2 >= THRESH);
      accv[r]   += fire ? 1.0f : 0.0f;
      mem2[r]    = fire ? 0.0f : m2;
    }
  }

  // ---------------- Store: lanes 0-3 / 16-19 hold distinct (row, action) ----
  if (((lane >> 2) & 3) == 0) {
    const float inv  = 1.0f / (float)T;
    const int   row0 = rowBase + ((lane >> 4) << 3);
    #pragma unroll
    for (int r = 0; r < 8; ++r)
      out[(size_t)(row0 + r) * 4 + asel] = accv[r] * inv;
  }
}

extern "C" void kernel_launch(void* const* d_in, const int* in_sizes, int n_in,
                              void* d_out, int out_size, void* d_ws, size_t ws_size,
                              hipStream_t stream) {
  const float* X  = (const float*)d_in[0];   // sensor_spikes [B,1024]
  const float* W1 = (const float*)d_in[1];   // [64,1024]
  const float* b1 = (const float*)d_in[2];   // [64]
  const float* W2 = (const float*)d_in[3];   // [4,64]
  const float* b2 = (const float*)d_in[4];   // [4]
  const int*   ts = (const int*)d_in[5];     // time_steps (device scalar)
  float* out = (float*)d_out;

  const int batch = in_sizes[0] / 1024;      // 65536
  const int grid  = batch / 128;             // 128 rows per 256-thread WG
  snn_fused_kernel<<<grid, 256, 0, stream>>>(X, W1, b1, W2, b2, ts, out);
}
